// Attention4DDownsample_72859825209947
// MI455X (gfx1250) — matrix operand order repaired and checked by reading, not compiled, not run on hardware
//
#include <hip/hip_runtime.h>
#include <hip/hip_bf16.h>
#include <math.h>

// ---------------------------------------------------------------------------
// Attention4DDownsample for MI455X (gfx1250, wave32, WMMA bf16 16x16x32).
// Memory-bound (~78 GFLOP vs ~0.6-0.9 GB HBM traffic @ 23.3 TB/s):
//  - bf16 WMMA for every GEMM contraction (halves operand bytes)
//  - BN folded into conv weights once
//  - bias gather / softmax / v_local dwconv / GELU fused per (b,h)
//  - double-buffered GEMM staging, paired independent WMMAs (fill hazards)
// ---------------------------------------------------------------------------

typedef __bf16 bf16_t;
typedef __attribute__((ext_vector_type(16))) __bf16          bf16x16;
typedef __attribute__((ext_vector_type(8)))  float           f32x8;
typedef __attribute__((ext_vector_type(8)))  unsigned short  u16x8;

union Frag {
    bf16x16 v;
    u16x8   h[2];
    unsigned short u[16];
};

static __device__ __forceinline__ u16x8 ld8(const bf16_t* p) {
    return *reinterpret_cast<const u16x8*>(p);
}

// Problem constants
#define BATCH   512
#define DIMC    384
#define RES     14
#define RES2    7
#define NPOS    196     // 14*14
#define N2POS   49      // 7*7
#define NHEADS  8
#define KDIM    16
#define DHEAD   64
#define DHC     512     // NHEADS*DHEAD
#define NHKD    128     // NHEADS*KDIM
#define OUTDIM  768
#define JP      224     // 196 padded to 7*32 (K chunks of 32)

// ---------------------------------------------------------------------------
// Kernel 1: fold BN into weights (bf16), fold biases, gather bias table.
// ---------------------------------------------------------------------------
__global__ void prep_kernel(
    const float* __restrict__ qw,  const float* __restrict__ qb,
    const float* __restrict__ qs,  const float* __restrict__ qt,
    const float* __restrict__ kw,  const float* __restrict__ kb,
    const float* __restrict__ ks,  const float* __restrict__ kt,
    const float* __restrict__ vw,  const float* __restrict__ vb,
    const float* __restrict__ vs,  const float* __restrict__ vt,
    const float* __restrict__ pw,  const float* __restrict__ pb,
    const float* __restrict__ ps,  const float* __restrict__ pt,
    const float* __restrict__ vlw, const float* __restrict__ vlb,
    const float* __restrict__ vls, const float* __restrict__ vlt,
    const float* __restrict__ abias, const int* __restrict__ bidx, int n_off,
    bf16_t* __restrict__ wq, bf16_t* __restrict__ wk,
    bf16_t* __restrict__ wv, bf16_t* __restrict__ wp,
    float* __restrict__ bq, float* __restrict__ bk,
    float* __restrict__ bv, float* __restrict__ bp,
    float* __restrict__ vlwf, float* __restrict__ vlbf,
    float* __restrict__ btab)
{
    int i = blockIdx.x * blockDim.x + threadIdx.x;
    if (i < NHKD*DIMC)   { int o = i / DIMC;  wq[i] = (bf16_t)(qw[i] * qs[o]); return; }
    i -= NHKD*DIMC;
    if (i < NHKD*DIMC)   { int o = i / DIMC;  wk[i] = (bf16_t)(kw[i] * ks[o]); return; }
    i -= NHKD*DIMC;
    if (i < DHC*DIMC)    { int o = i / DIMC;  wv[i] = (bf16_t)(vw[i] * vs[o]); return; }
    i -= DHC*DIMC;
    if (i < OUTDIM*DHC)  { int o = i / DHC;   wp[i] = (bf16_t)(pw[i] * ps[o]); return; }
    i -= OUTDIM*DHC;
    if (i < DHC*9)       { int c = i / 9;     vlwf[i] = vlw[i] * vls[c];       return; }
    i -= DHC*9;
    if (i < NHKD)        { bq[i]   = qb[i]*qs[i]   + qt[i];  return; }  i -= NHKD;
    if (i < NHKD)        { bk[i]   = kb[i]*ks[i]   + kt[i];  return; }  i -= NHKD;
    if (i < DHC)         { bv[i]   = vb[i]*vs[i]   + vt[i];  return; }  i -= DHC;
    if (i < OUTDIM)      { bp[i]   = pb[i]*ps[i]   + pt[i];  return; }  i -= OUTDIM;
    if (i < DHC)         { vlbf[i] = vlb[i]*vls[i] + vlt[i]; return; }  i -= DHC;
    if (i < NHEADS*N2POS*NPOS) {
        int h = i / (N2POS*NPOS);
        btab[i] = abias[h * n_off + bidx[i % (N2POS*NPOS)]];
    }
}
#define PREP_TOTAL (NHKD*DIMC + NHKD*DIMC + DHC*DIMC + OUTDIM*DHC + DHC*9 \
                    + NHKD + NHKD + DHC + OUTDIM + DHC + NHEADS*N2POS*NPOS)

// ---------------------------------------------------------------------------
// Kernel 2: ql = dwconv3x3 stride2 (x) + x[:, :, ::2, ::2]   (f32, [b][c][49])
// ---------------------------------------------------------------------------
__global__ void ql_kernel(const float* __restrict__ x,
                          const float* __restrict__ w,
                          const float* __restrict__ bias,
                          float* __restrict__ ql)
{
    int i = blockIdx.x * blockDim.x + threadIdx.x;
    if (i >= BATCH*DIMC*N2POS) return;
    int n2 = i % N2POS;
    int c  = (i / N2POS) % DIMC;
    int b  = i / (N2POS*DIMC);
    int yo = n2 / RES2, xo = n2 % RES2;
    const float* xp = x + ((size_t)b*DIMC + c) * NPOS;
    float acc = bias[c] + xp[(2*yo)*RES + 2*xo];   // strided residual
    #pragma unroll
    for (int dy = 0; dy < 3; ++dy) {
        int y = 2*yo + dy - 1;
        if ((unsigned)y >= (unsigned)RES) continue;
        #pragma unroll
        for (int dx = 0; dx < 3; ++dx) {
            int xx = 2*xo + dx - 1;
            if ((unsigned)xx >= (unsigned)RES) continue;
            acc += w[c*9 + dy*3 + dx] * xp[y*RES + xx];
        }
    }
    ql[i] = acc;
}

// ---------------------------------------------------------------------------
// Kernel 3: generic 1x1-conv GEMM, C[b][m][n] = sum_k W[m][k]*X[b][k][n]+bias.
// 64x64 tile per 256-thread block (8 waves, 2 wmma tiles each), K chunk 32,
// double-buffered LDS staging (global loads overlap WMMA compute).
// MODE 0: f32 out [b][M][N].  MODE 1: bf16 out transposed [b][h][n][m&15].
// ---------------------------------------------------------------------------
#define GBUF (64*40)    // elements per LDS buffer (stride 40 avoids conflicts)

template<int MODE, bool XBF>
__global__ __launch_bounds__(256) void gemm1x1_kernel(
    const float* __restrict__ Xf, const bf16_t* __restrict__ Xb,
    const bf16_t* __restrict__ W, const float* __restrict__ bias,
    float* __restrict__ outF, bf16_t* __restrict__ outT,
    int M, int N, int K)
{
    const int b  = blockIdx.z;
    const int m0 = blockIdx.y * 64;
    const int n0 = blockIdx.x * 64;
    __shared__ __align__(16) bf16_t wl[2*GBUF];    // [buf][m][k]
    __shared__ __align__(16) bf16_t xl[2*GBUF];    // [buf][n][k]
    const int tid  = threadIdx.x;
    const int lane = tid & 31, wave = tid >> 5;
    const int msub = wave & 3, ngrp = wave >> 2;
    const int nc = lane & 15, half = lane >> 4;
    f32x8 acc0 = {}, acc1 = {};
    const size_t xbase = (size_t)b * K * N;
    const bool fullN = (n0 + 64 <= N);
    const int kk = tid & 31, ng = tid >> 5;         // X staging coords
    const int wrow = tid >> 2, wkg = (tid & 3) * 8; // W staging coords

    auto stage_load = [&](int k0, u16x8& wr, float (&x8)[8]) {
        wr = *reinterpret_cast<const u16x8*>(W + (size_t)(m0 + wrow)*K + k0 + wkg);
        if (!XBF && fullN) {
            const float* src = Xf + xbase + (size_t)(k0 + kk) * N + n0 + ng*8;
            float4 p0 = *reinterpret_cast<const float4*>(src);
            float4 p1 = *reinterpret_cast<const float4*>(src + 4);
            x8[0]=p0.x; x8[1]=p0.y; x8[2]=p0.z; x8[3]=p0.w;
            x8[4]=p1.x; x8[5]=p1.y; x8[6]=p1.z; x8[7]=p1.w;
        } else {
            #pragma unroll
            for (int ii = 0; ii < 8; ++ii) {
                int n = n0 + ng*8 + ii;
                float v = 0.f;
                if (n < N) {
                    size_t idx = xbase + (size_t)(k0 + kk) * N + n;
                    v = XBF ? (float)Xb[idx] : Xf[idx];
                }
                x8[ii] = v;
            }
        }
    };
    auto stage_store = [&](int buf, const u16x8& wr, const float (&x8)[8]) {
        *reinterpret_cast<u16x8*>(wl + buf*GBUF + wrow*40 + wkg) = wr;
        #pragma unroll
        for (int ii = 0; ii < 8; ++ii)
            xl[buf*GBUF + (ng*8 + ii)*40 + kk] = (bf16_t)x8[ii];
    };
    auto compute = [&](int buf) {
        const bf16_t* wb = wl + buf*GBUF;
        const bf16_t* xb = xl + buf*GBUF;
        Frag a, b0, b1;
        int m = msub*16 + nc;
        a.h[0] = ld8(wb + m*40 + half*8);           // e[0..7]=K(half*8..)
        a.h[1] = ld8(wb + m*40 + 16 + half*8);      // e[8..15]=K(16+half*8..)
        int r0 = ngrp*32 + nc;
        b0.h[0] = ld8(xb + r0*40 + half*8);         // e[0..7]=K(half*8..)
        b0.h[1] = ld8(xb + r0*40 + 16 + half*8);    // e[8..15]=K(16+half*8..)
        int r1 = r0 + 16;
        b1.h[0] = ld8(xb + r1*40 + half*8);
        b1.h[1] = ld8(xb + r1*40 + 16 + half*8);
        acc0 = __builtin_amdgcn_wmma_f32_16x16x32_bf16(false, a.v, false, b0.v,
                                                       (short)0, acc0, false, false);
        acc1 = __builtin_amdgcn_wmma_f32_16x16x32_bf16(false, a.v, false, b1.v,
                                                       (short)0, acc1, false, false);
    };

    const int nch = K >> 5;
    {   // prologue: fill buffer 0
        u16x8 wr; float x8[8];
        stage_load(0, wr, x8);
        stage_store(0, wr, x8);
    }
    __syncthreads();
    for (int i = 0; i < nch; ++i) {
        u16x8 wn; float xn[8];
        if (i + 1 < nch) stage_load((i + 1) << 5, wn, xn);   // loads in flight
        compute(i & 1);                                      // WMMA on current
        if (i + 1 < nch) stage_store((i + 1) & 1, wn, xn);   // fill other buf
        __syncthreads();
    }

    // epilogue: C layout -> lane col = lane&15, row = r + half*8
    float brow[8];
    #pragma unroll
    for (int r = 0; r < 8; ++r)
        brow[r] = bias[m0 + msub*16 + half*8 + r];
    #pragma unroll
    for (int t = 0; t < 2; ++t) {
        f32x8 acc = t ? acc1 : acc0;
        int n = n0 + ngrp*32 + t*16 + nc;
        if (n >= N) continue;
        #pragma unroll
        for (int r = 0; r < 8; ++r) {
            int m = m0 + msub*16 + half*8 + r;
            float val = acc[r] + brow[r];
            if (MODE == 0) {
                outF[(size_t)b*M*N + (size_t)m*N + n] = val;
            } else {
                int hh = m >> 4, cc = m & 15;
                outT[(((size_t)b*(M>>4) + hh)*N + n)*16 + cc] = (bf16_t)val;
            }
        }
    }
}

// ---------------------------------------------------------------------------
// Kernel 4: fused attention per (b,h): QK^T*scale+bias -> softmax ->
//           v_local dwconv -> probs@V -> +v_local -> exact GELU -> bf16 out.
// ---------------------------------------------------------------------------
__global__ __launch_bounds__(256) void attn_kernel(
    const bf16_t* __restrict__ qt, const bf16_t* __restrict__ ktr,
    const float*  __restrict__ v4, const float*  __restrict__ btab,
    const float*  __restrict__ vlw, const float* __restrict__ vlb,
    bf16_t* __restrict__ g)
{
    const int h = blockIdx.x;
    const int b = blockIdx.y;
    extern __shared__ char smem[];
    bf16_t* qlds = (bf16_t*)smem;               // [64][16]
    bf16_t* klds = qlds + 64*16;                // [224][32]  upper K half zeroed
    bf16_t* vlds = klds + JP*32;                // [64][224]  (d-major, j contiguous)
    float*  slds = (float*)(vlds + 64*JP);      // [64][224]  bias -> logits f32
    bf16_t* plds = (bf16_t*)(slds + 64*JP);     // [64][224]  probs bf16
    float*  vll  = (float*)(plds + 64*JP);      // [64][49]   v_local
    const int tid = threadIdx.x, lane = tid & 31, wave = tid >> 5;
    const int nc = lane & 15, half = lane >> 4;

    // ---- setup (coalesced, shift-indexed)
    {
        int c = tid & 15, r0 = tid >> 4;             // 16 rows in flight
        for (int r = r0; r < 64; r += 16)
            qlds[r*16 + c] = (r < N2POS)
                ? qt[(((size_t)b*NHEADS + h)*N2POS + r)*16 + c] : (bf16_t)0.f;
        for (int r = r0; r < JP; r += 16) {          // K 0..15 data, 16..31 zero
            klds[r*32 + c]      = (r < NPOS)
                ? ktr[(((size_t)b*NHEADS + h)*NPOS + r)*16 + c] : (bf16_t)0.f;
            klds[r*32 + 16 + c] = (bf16_t)0.f;
        }
    }
    for (int d = wave; d < DHEAD; d += 8) {
        const float* vp = v4 + ((size_t)b*DHC + h*DHEAD + d)*NPOS;
        for (int j = lane; j < JP; j += 32)
            vlds[d*JP + j] = (j < NPOS) ? (bf16_t)vp[j] : (bf16_t)0.f;
    }
    for (int qi = wave; qi < 64; qi += 8) {          // prefill logits with bias
        const float* bp = btab + (h*N2POS + qi)*NPOS;
        for (int j = lane; j < JP; j += 32)
            slds[qi*JP + j] = (qi < N2POS && j < NPOS) ? bp[j] : -1e30f;
    }
    __syncthreads();

    // ---- GEMM1: S = q(49x16) @ k(16x196); wave -> fixed M tile, 7 N tiles.
    {
        const int mt = wave >> 1;                    // 0..3
        const int nb = (wave & 1) * 7;               // nt = nb + i, i in 0..6
        Frag a = {};                                 // shared A, K16..31 zero
        a.h[0] = ld8(qlds + (mt*16 + nc)*16 + half*8);
        const int sbase = (mt*16 + half*8)*JP + nc;
        auto epi = [&](const f32x8& c, int nt) {     // logits = 0.25*S + bias
            #pragma unroll
            for (int r = 0; r < 8; ++r) {
                int idx = sbase + r*JP + nt*16;
                slds[idx] = fmaf(c[r], 0.25f, slds[idx]);
            }
        };
        #pragma unroll
        for (int ip = 0; ip < 3; ++ip) {             // paired independent WMMAs
            int nt0 = nb + ip*2, nt1 = nt0 + 1;
            Frag b0, b1;
            b0.h[0] = ld8(klds + (nt0*16 + nc)*32 + half*8);
            b0.h[1] = ld8(klds + (nt0*16 + nc)*32 + 16 + half*8);
            b1.h[0] = ld8(klds + (nt1*16 + nc)*32 + half*8);
            b1.h[1] = ld8(klds + (nt1*16 + nc)*32 + 16 + half*8);
            f32x8 c0 = {}, c1 = {};
            c0 = __builtin_amdgcn_wmma_f32_16x16x32_bf16(false, a.v, false, b0.v,
                                                         (short)0, c0, false, false);
            c1 = __builtin_amdgcn_wmma_f32_16x16x32_bf16(false, a.v, false, b1.v,
                                                         (short)0, c1, false, false);
            epi(c0, nt0);
            epi(c1, nt1);
        }
        {   // last N tile
            int nt = nb + 6;
            Frag bb;
            bb.h[0] = ld8(klds + (nt*16 + nc)*32 + half*8);
            bb.h[1] = ld8(klds + (nt*16 + nc)*32 + 16 + half*8);
            f32x8 c = {};
            c = __builtin_amdgcn_wmma_f32_16x16x32_bf16(false, a.v, false, bb.v,
                                                        (short)0, c, false, false);
            epi(c, nt);
        }
    }
    __syncthreads();

    // ---- softmax: one warp per row, rows 0..63 (padded rows are benign)
    for (int rr = 0; rr < 8; ++rr) {
        int row = rr * 8 + wave;
        float mx = -1e30f;
        #pragma unroll
        for (int jj = 0; jj < 7; ++jj)
            mx = fmaxf(mx, slds[row*JP + lane + jj*32]);
        for (int o = 16; o > 0; o >>= 1) mx = fmaxf(mx, __shfl_xor(mx, o));
        float e[7], sum = 0.f;
        #pragma unroll
        for (int jj = 0; jj < 7; ++jj) {
            e[jj] = expf(slds[row*JP + lane + jj*32] - mx);
            sum += e[jj];
        }
        for (int o = 16; o > 0; o >>= 1) sum += __shfl_xor(sum, o);
        float inv = 1.f / sum;
        #pragma unroll
        for (int jj = 0; jj < 7; ++jj)
            plds[row*JP + lane + jj*32] = (bf16_t)(e[jj] * inv);
    }

    // ---- v_local: 3x3 stride-2 dwconv on the LDS-resident v tile
    for (int i = tid; i < DHEAD*N2POS; i += 256) {
        int d = i / N2POS, n2 = i % N2POS;
        int yo = n2 / RES2, xo = n2 % RES2;
        int ch = h*DHEAD + d;
        float acc = vlb[ch];
        #pragma unroll
        for (int dy = 0; dy < 3; ++dy) {
            int y = 2*yo + dy - 1;
            if ((unsigned)y >= (unsigned)RES) continue;
            #pragma unroll
            for (int dx = 0; dx < 3; ++dx) {
                int xx = 2*xo + dx - 1;
                if ((unsigned)xx >= (unsigned)RES) continue;
                acc += vlw[ch*9 + dy*3 + dx] * (float)vlds[d*JP + y*RES + xx];
            }
        }
        vll[i] = acc;
    }
    __syncthreads();

    // ---- GEMM2: out(49x64) = probs(49x224) @ v(224x64); wave -> fixed M tile,
    //      two N tiles, shared A per kc chunk, paired independent WMMAs.
    {
        const int mt = wave >> 1;                    // 0..3
        const int nt0 = (wave & 1) * 2, nt1 = nt0 + 1;
        const int m = mt*16 + nc;
        const int d0 = nt0*16 + nc, d1 = d0 + 16;
        f32x8 c0 = {}, c1 = {};
        #pragma unroll
        for (int kc = 0; kc < 7; ++kc) {
            Frag a, b0, b1;
            a.h[0] = ld8(plds + m*JP + kc*32 + half*8);
            a.h[1] = ld8(plds + m*JP + kc*32 + 16 + half*8);
            b0.h[0] = ld8(vlds + d0*JP + kc*32 + half*8);
            b0.h[1] = ld8(vlds + d0*JP + kc*32 + 16 + half*8);
            b1.h[0] = ld8(vlds + d1*JP + kc*32 + half*8);
            b1.h[1] = ld8(vlds + d1*JP + kc*32 + 16 + half*8);
            c0 = __builtin_amdgcn_wmma_f32_16x16x32_bf16(false, a.v, false, b0.v,
                                                         (short)0, c0, false, false);
            c1 = __builtin_amdgcn_wmma_f32_16x16x32_bf16(false, a.v, false, b1.v,
                                                         (short)0, c1, false, false);
        }
        #pragma unroll
        for (int t = 0; t < 2; ++t) {
            const f32x8& c = t ? c1 : c0;
            int d = (t ? nt1 : nt0)*16 + nc;
            #pragma unroll
            for (int r = 0; r < 8; ++r) {
                int qi = mt*16 + half*8 + r;
                if (qi < N2POS) {
                    float val = c[r] + vll[d*N2POS + qi];
                    float gv  = 0.5f * val * (1.f + erff(val * 0.70710678118654752f));
                    g[((size_t)b*DHC + h*DHEAD + d)*N2POS + qi] = (bf16_t)gv;
                }
            }
        }
    }
}

// ---------------------------------------------------------------------------
// Host launcher
// ---------------------------------------------------------------------------
extern "C" void kernel_launch(void* const* d_in, const int* in_sizes, int n_in,
                              void* d_out, int out_size, void* d_ws, size_t ws_size,
                              hipStream_t stream)
{
    const float* x     = (const float*)d_in[0];
    const float* qlw   = (const float*)d_in[1];
    const float* qlb   = (const float*)d_in[2];
    const float* qpw   = (const float*)d_in[3];
    const float* qpb   = (const float*)d_in[4];
    const float* qbs   = (const float*)d_in[5];
    const float* qbt   = (const float*)d_in[6];
    const float* kw    = (const float*)d_in[7];
    const float* kb    = (const float*)d_in[8];
    const float* kbs   = (const float*)d_in[9];
    const float* kbt   = (const float*)d_in[10];
    const float* vw    = (const float*)d_in[11];
    const float* vb    = (const float*)d_in[12];
    const float* vbs   = (const float*)d_in[13];
    const float* vbt   = (const float*)d_in[14];
    const float* vlw   = (const float*)d_in[15];
    const float* vlb   = (const float*)d_in[16];
    const float* vlbs  = (const float*)d_in[17];
    const float* vlbt  = (const float*)d_in[18];
    const float* pw    = (const float*)d_in[19];
    const float* pb    = (const float*)d_in[20];
    const float* pbs   = (const float*)d_in[21];
    const float* pbt   = (const float*)d_in[22];
    const float* abias = (const float*)d_in[23];
    const int*   bidx  = (const int*)d_in[24];
    const int n_off = in_sizes[23] / NHEADS;
    float* out = (float*)d_out;

    char* ws = (char*)d_ws;
    size_t off = 0;
    auto alloc = [&](size_t bytes) -> void* {
        void* p = ws + off;
        off = (off + bytes + 255) & ~(size_t)255;
        return p;
    };
    bf16_t* wq  = (bf16_t*)alloc((size_t)NHKD*DIMC*2);
    bf16_t* wk  = (bf16_t*)alloc((size_t)NHKD*DIMC*2);
    bf16_t* wv  = (bf16_t*)alloc((size_t)DHC*DIMC*2);
    bf16_t* wp  = (bf16_t*)alloc((size_t)OUTDIM*DHC*2);
    float*  bq  = (float*) alloc(NHKD*4);
    float*  bk  = (float*) alloc(NHKD*4);
    float*  bv  = (float*) alloc(DHC*4);
    float*  bp  = (float*) alloc(OUTDIM*4);
    float*  vlwf= (float*) alloc(DHC*9*4);
    float*  vlbf= (float*) alloc(DHC*4);
    float*  btab= (float*) alloc((size_t)NHEADS*N2POS*NPOS*4);
    float*  qlbuf=(float*) alloc((size_t)BATCH*DIMC*N2POS*4);
    float*  v4  = (float*) alloc((size_t)BATCH*DHC*NPOS*4);
    bf16_t* ktr = (bf16_t*)alloc((size_t)BATCH*NHEADS*NPOS*KDIM*2);
    bf16_t* qtr = (bf16_t*)alloc((size_t)BATCH*NHEADS*N2POS*KDIM*2);
    bf16_t* g   = (bf16_t*)alloc((size_t)BATCH*DHC*N2POS*2);
    (void)ws_size; (void)n_in; (void)out_size;

    // 1. fold BN + gather bias table
    prep_kernel<<<(PREP_TOTAL + 255)/256, 256, 0, stream>>>(
        qpw, qpb, qbs, qbt, kw, kb, kbs, kbt, vw, vb, vbs, vbt,
        pw, pb, pbs, pbt, vlw, vlb, vlbs, vlbt, abias, bidx, n_off,
        wq, wk, wv, wp, bq, bk, bv, bp, vlwf, vlbf, btab);

    // 2. ql = dwconv(x) + x[::2,::2]
    ql_kernel<<<(BATCH*DIMC*N2POS + 255)/256, 256, 0, stream>>>(x, qlw, qlb, qlbuf);

    // 3. v projection: [b][512][196] f32
    gemm1x1_kernel<0, false><<<dim3(4, DHC/64, BATCH), 256, 0, stream>>>(
        x, nullptr, wv, bv, v4, nullptr, DHC, NPOS, DIMC);
    // 4. k projection -> head-transposed bf16 [b][h][196][16]
    gemm1x1_kernel<1, false><<<dim3(4, NHKD/64, BATCH), 256, 0, stream>>>(
        x, nullptr, wk, bk, nullptr, ktr, NHKD, NPOS, DIMC);
    // 5. q projection -> head-transposed bf16 [b][h][49][16]
    gemm1x1_kernel<1, false><<<dim3(1, NHKD/64, BATCH), 256, 0, stream>>>(
        qlbuf, nullptr, wq, bq, nullptr, qtr, NHKD, N2POS, DIMC);

    // 6. fused attention + v_local + GELU  (per (head, batch))
    size_t smem = (size_t)64*16*2          // q bf16
                + (size_t)JP*32*2          // k bf16 (K padded to 32)
                + (size_t)64*JP*2          // v bf16
                + (size_t)64*JP*4          // logits f32
                + (size_t)64*JP*2          // probs bf16
                + (size_t)DHEAD*N2POS*4;   // v_local f32
    attn_kernel<<<dim3(NHEADS, BATCH), 256, smem, stream>>>(
        qtr, ktr, v4, btab, vlwf, vlbf, g);

    // 7. output projection: [b][768][49] f32 -> d_out
    gemm1x1_kernel<0, true><<<dim3(1, OUTDIM/64, BATCH), 256, 0, stream>>>(
        nullptr, g, wp, bp, out, nullptr, OUTDIM, N2POS, DHC);
}
